// sequential_bi_tied_36344013258824
// MI455X (gfx1250) — matrix operand order repaired and checked by reading, not compiled, not run on hardware
//
#include <hip/hip_runtime.h>
#include <hip/hip_bf16.h>

// ---------------------------------------------------------------------------
// Bidirectional tied-kernel long convolution as a triangular Toeplitz GEMM.
//   y[b,h,i] = sum_{j<=i} x[b,h,j] k[h,i-j] + sum_{j>=i} x[b,h,j] k[h,j-i]
//            + D[h] * x[b,h,i]
// Per head h:  U = C_h * X_h,  C_h[i,j] = k[h,i-j] (lower-tri Toeplitz),
// X_h = [x_0..x_7 | rev(x_0)..rev(x_7)]  (L x 16).
//   y[b,h,i] = U[i,b] + U[L-1-i, b+8] + D[h]*x[b,h,i]
// 275 GFLOP -> f16 WMMA w/ f32 accum; HBM floor ~12us, WMMA ~100us dominates.
// x/k staged in LDS once per head; A-tiles are Toeplitz and are gathered from
// a reversed kernel copy (zero tail == lower-triangular mask). Two parity-
// shifted krev copies make every A-fragment run 4B-aligned (ds_load_b32).
// ---------------------------------------------------------------------------

typedef __attribute__((ext_vector_type(16))) _Float16 v16h;
typedef __attribute__((ext_vector_type(8)))  float    v8f;

#define B_       8
#define H_       1024
#define L_       4096
#define XSTRIDE  4120          // LDS col stride (elems): mult of 8, >= 4112, bank-staggered
#define KREV_N   4128          // 4096 + 32 zero pad (implements d<0 mask)
#define NTHREADS 512
#define NWAVES   (NTHREADS / 32)

union AFrag  { uint u[8];       v16h v; };
union BFrag  { uint4 u[2];      v16h v; };
union Pack4  { _Float16 h[4];   uint u2[2]; };

__global__ __launch_bounds__(NTHREADS, 1)
void bi_tied_conv_wmma(const float* __restrict__ X,
                       const float* __restrict__ K,
                       const float* __restrict__ Dp,
                       float* __restrict__ Y) {
  // 16 cols x 4120 rows f16 = 131840 B + 2 x 8256 B krev copies  (~148 KB)
  __shared__ __align__(16) _Float16 lds_x[16 * XSTRIDE];
  __shared__ __align__(16) _Float16 lds_krev0[KREV_N];  // krev0[s] = k[h,4095-s]
  __shared__ __align__(16) _Float16 lds_krev1[KREV_N];  // krev1[s] = krev0[s+1]

  const int h   = blockIdx.x;
  const int tid = threadIdx.x;

  // ---- Stage X_h: cols 0-7 forward, 8-15 reversed (f32->f16, vectorized) --
  for (int idx = tid; idx < (B_ * L_) / 4; idx += NTHREADS) {
    const int b  = idx >> 10;             // / (L_/4)
    const int j0 = (idx & ((L_ / 4) - 1)) * 4;
    const float4 xv = *(const float4*)(X + ((size_t)(b * H_ + h)) * (size_t)L_ + j0);
    Pack4 f, r;
    f.h[0] = (_Float16)xv.x; f.h[1] = (_Float16)xv.y;
    f.h[2] = (_Float16)xv.z; f.h[3] = (_Float16)xv.w;
    r.h[0] = f.h[3]; r.h[1] = f.h[2]; r.h[2] = f.h[1]; r.h[3] = f.h[0];
    *(uint2*)(&lds_x[b * XSTRIDE + j0]) = *(const uint2*)f.u2;              // ds_store_b64
    *(uint2*)(&lds_x[(b + 8) * XSTRIDE + (L_ - 4 - j0)]) = *(const uint2*)r.u2;
  }
  // Zero pad rows (L_..XSTRIDE-1 of every column): kills OOB-j reads.
  for (int idx = tid; idx < 16 * (XSTRIDE - L_); idx += NTHREADS) {
    const int c = idx / (XSTRIDE - L_);
    const int j = L_ + idx % (XSTRIDE - L_);
    lds_x[c * XSTRIDE + j] = (_Float16)0.f;
  }
  // Reversed kernel, two parity-shifted copies; zero tails give the
  // lower-triangular mask for free (d<0 -> s>4095 -> 0).
  for (int s = tid; s < KREV_N; s += NTHREADS) {
    const float v0 = (s < L_)       ? K[(size_t)h * (size_t)L_ + (L_ - 1 - s)] : 0.f;
    const float v1 = ((s + 1) < L_) ? K[(size_t)h * (size_t)L_ + (L_ - 2 - s)] : 0.f;
    lds_krev0[s] = (_Float16)v0;
    lds_krev1[s] = (_Float16)v1;
  }
  __syncthreads();

  // Per-lane WMMA fragment geometry (wave32, 16x16x32 f16 per ISA 7.12.2):
  //  A 16x32:  lanes 0-15 -> M=lane, K in {0..7,16..23}; lanes 16-31 -> K +8
  //  B 32x16:  lanes 0-15 -> N=lane, K in {0..7,16..23}; lanes 16-31 -> K +8
  //  C 16x16:  VGPR v -> M = v + 8*(lane>=16), N = lane&15
  const int lane  = tid & 31;
  const int wave  = tid >> 5;
  const int half  = lane >> 4;
  const int n     = lane & 15;
  const int kbase = half * 8;            // A K-base
  const int khalf = half * 8;            // B K-base
  const int a0c   = 4095 + kbase - n;    // s0 = a0c - 16*(I-J); parity fixed/lane
  const float Dh  = Dp[h];
  // Parity-selected base: kbasep[s0] == krev[s0] with s0-kbasep 4B-aligned.
  const _Float16* const kbasep = (a0c & 1) ? (lds_krev1 - 1) : lds_krev0;
  const _Float16* const bcol   = &lds_x[n * XSTRIDE + khalf];

  for (int p = 0; p < 8; ++p) {
    const int I0 = wave + 16 * p;        // 0..127
    const int I1 = 255 - I0;             // 128..255 (pair => constant 257 blocks)
    v8f acc0 = {0.f, 0.f, 0.f, 0.f, 0.f, 0.f, 0.f, 0.f};
    v8f acc1 = {0.f, 0.f, 0.f, 0.f, 0.f, 0.f, 0.f, 0.f};

    // ---- acc0 = row-block I0 of C_h * X_h -------------------------------
    {
      const _Float16* kq = kbasep + (a0c - 16 * I0);
      const _Float16* bp = bcol;
      for (int J = 0; J <= I0; J += 2) {
        BFrag bf;
        bf.u[0] = *(const uint4*)(bp);       // ds_load_b128
        bf.u[1] = *(const uint4*)(bp + 16);  // ds_load_b128
        AFrag af;
        const uint* qa = (const uint*)kq;    // 4B-aligned by construction
#pragma unroll
        for (int e = 0; e < 4; ++e) { af.u[e] = qa[e]; af.u[4 + e] = qa[8 + e]; }
        acc0 = __builtin_amdgcn_wmma_f32_16x16x32_f16(
            false, af.v, false, bf.v, (short)0, acc0, false, false);
        kq += 32; bp += 32;
      }
    }
    // ---- acc1 = row-block I1 --------------------------------------------
    {
      const _Float16* kq = kbasep + (a0c - 16 * I1);
      const _Float16* bp = bcol;
      for (int J = 0; J <= I1; J += 2) {
        BFrag bf;
        bf.u[0] = *(const uint4*)(bp);
        bf.u[1] = *(const uint4*)(bp + 16);
        AFrag af;
        const uint* qa = (const uint*)kq;
#pragma unroll
        for (int e = 0; e < 4; ++e) { af.u[e] = qa[e]; af.u[4 + e] = qa[8 + e]; }
        acc1 = __builtin_amdgcn_wmma_f32_16x16x32_f16(
            false, af.v, false, bf.v, (short)0, acc1, false, false);
        kq += 32; bp += 32;
      }
    }

    // ---- Combine: y[I0-blk] = acc0[:,0:8] + flip(acc1)[:,8:16]
    //              y[I1-blk] = acc1[:,0:8] + flip(acc0)[:,8:16]
    // acc?[15-m, n+8] lives at lane (lane+24)&31, VGPR 7-v.
    const int src = (lane + 24) & 31;
#pragma unroll
    for (int v = 0; v < 8; ++v) {
      const float pA = __shfl(acc1[7 - v], src, 32);
      const float pB = __shfl(acc0[7 - v], src, 32);
      const float oA = acc0[v] + pA;
      const float oB = acc1[v] + pB;
      if (n < 8) {                        // cols 0-7 are the real batches
        const int m  = v + 8 * half;
        const int iA = I0 * 16 + m;
        const int iB = I1 * 16 + m;
        const size_t base = ((size_t)(n * H_ + h)) * (size_t)L_;
        Y[base + iA] = oA + Dh * (float)lds_x[n * XSTRIDE + iA];
        Y[base + iB] = oB + Dh * (float)lds_x[n * XSTRIDE + iB];
      }
    }
  }
}

extern "C" void kernel_launch(void* const* d_in, const int* in_sizes, int n_in,
                              void* d_out, int out_size, void* d_ws, size_t ws_size,
                              hipStream_t stream) {
  const float* x = (const float*)d_in[0];   // (8, 1024, 4096) f32
  const float* k = (const float*)d_in[1];   // (1, 1024, 4096) f32
  const float* D = (const float*)d_in[2];   // (1, 1024, 1)    f32
  float* y = (float*)d_out;                 // (8, 1024, 4096) f32
  (void)in_sizes; (void)n_in; (void)out_size; (void)d_ws; (void)ws_size;

  dim3 grid(H_);
  dim3 block(NTHREADS);
  hipLaunchKernelGGL(bi_tied_conv_wmma, grid, block, 0, stream, x, k, D, y);
}
